// GATv2Model_30116310679910
// MI455X (gfx1250) — matrix operand order repaired and checked by reading, not compiled, not run on hardware
//
#include <hip/hip_runtime.h>
#include <hip/hip_bf16.h>
#include <hip/hip_fp16.h>

// ---- problem constants (match reference) ----
#define N_NODES 10000
#define N_EDGES 320000
#define F_INP   7
#define DIM     256
#define HEADS   8
#define CHAN    32
#define GRAPHS  16
#define SLOPE   0.2f

typedef __attribute__((ext_vector_type(16))) _Float16 v16h;
typedef __attribute__((ext_vector_type(8)))  _Float16 v8h;
typedef __attribute__((ext_vector_type(8)))  float    v8f;

// float atomic max via signed/unsigned integer ordering trick (init to -inf)
__device__ __forceinline__ void atomic_max_f32(float* addr, float v) {
    if (v >= 0.0f) atomicMax((int*)addr, __float_as_int(v));
    else           atomicMin((unsigned int*)addr, __float_as_uint(v));
}

// ---------------------------------------------------------------------------
// 0) Weight prep: Wl/Wr [L][K=256][N=256] f32 -> f16, transposed to [N][K]
//    so WMMA B-fragments become contiguous 16B loads.
// ---------------------------------------------------------------------------
__global__ void prep_weights(const float* __restrict__ Wl,
                             const float* __restrict__ Wr,
                             _Float16* __restrict__ Wt) {
    int idx = blockIdx.x * blockDim.x + threadIdx.x;     // 4 * 65536
    if (idx >= 4 * DIM * DIM) return;
    int mat = idx >> 16;            // 0: l0 Wl, 1: l0 Wr, 2: l1 Wl, 3: l1 Wr
    int rem = idx & 0xFFFF;
    int n = rem >> 8, k = rem & 255;
    int l = mat >> 1;
    const float* W = (mat & 1) ? Wr : Wl;
    Wt[mat * DIM * DIM + n * DIM + k] =
        (_Float16)W[l * DIM * DIM + k * DIM + n];
}

// ---------------------------------------------------------------------------
// 1) Input projection: h = x @ Wp + bp   (K=7, scalar FMA; also emit f16 copy)
// ---------------------------------------------------------------------------
__global__ void project(const float* __restrict__ x,
                        const float* __restrict__ Wp,
                        const float* __restrict__ bp,
                        float* __restrict__ h,
                        _Float16* __restrict__ hf) {
    int idx = blockIdx.x * blockDim.x + threadIdx.x;     // N * D
    if (idx >= N_NODES * DIM) return;
    int n = idx >> 8, d = idx & 255;
    float acc = bp[d];
    #pragma unroll
    for (int k = 0; k < F_INP; k++)
        acc += x[n * F_INP + k] * Wp[k * DIM + d];
    h[idx]  = acc;
    hf[idx] = (_Float16)acc;
}

// ---------------------------------------------------------------------------
// 2) Per-layer init: acc = 0, denom = 0, m = -inf
// ---------------------------------------------------------------------------
__global__ void layer_init(float* __restrict__ acc,
                           float* __restrict__ denom,
                           float* __restrict__ m) {
    int idx = blockIdx.x * blockDim.x + threadIdx.x;
    if (idx < N_NODES * DIM) acc[idx] = 0.0f;
    if (idx < N_NODES * HEADS) {
        denom[idx] = 0.0f;
        m[idx] = -__builtin_huge_valf();
    }
}

// ---------------------------------------------------------------------------
// 3) WMMA GEMM: O = A(hf16 [10000,256]) x Bt(f16 [256,256], [n][k] layout)
//    One 16x16 output tile per wave, K swept in 8 steps of 32.
//    blockIdx.z selects (Wl -> xl) vs (Wr -> xr).
//    Grid (625, 2, 2), block 256 (= 8 wave32).
// ---------------------------------------------------------------------------
__global__ void gemm_wmma(const _Float16* __restrict__ A,
                          const _Float16* __restrict__ Bt0,
                          const _Float16* __restrict__ Bt1,
                          float* __restrict__ O0,
                          float* __restrict__ O1) {
    const _Float16* Bt = blockIdx.z ? Bt1 : Bt0;
    float* O = blockIdx.z ? O1 : O0;

    const int lane  = threadIdx.x & 31;
    const int wave  = threadIdx.x >> 5;
    const int tileM = blockIdx.x;                 // 0..624
    const int tileN = blockIdx.y * 8 + wave;      // 0..15
    const int l16   = lane & 15;
    const int hi    = lane >> 4;                  // 0 or 1

    // A fragment: row = tileM*16 + lane%16; halves {hi*8..+7, 16+hi*8..+7}
    const _Float16* aRow = A + (tileM * 16 + l16) * DIM + hi * 8;
    // B fragment: col = tileN*16 + lane%16; halves {hi*8..+7, 16+hi*8..+7}
    const _Float16* bRow = Bt + (tileN * 16 + l16) * DIM + hi * 8;

    v8f c = {};
    #pragma unroll
    for (int kk = 0; kk < DIM; kk += 32) {
        union { v16h v; v8h h[2]; } ua, ub;
        ua.h[0] = *(const v8h*)(aRow + kk);
        ua.h[1] = *(const v8h*)(aRow + kk + 16);
        ub.h[0] = *(const v8h*)(bRow + kk);
        ub.h[1] = *(const v8h*)(bRow + kk + 16);
        c = __builtin_amdgcn_wmma_f32_16x16x32_f16(
                /*neg_a=*/false, ua.v, /*neg_b=*/false, ub.v,
                /*c_mod=*/(short)0, c, /*reuse_a=*/false, /*reuse_b=*/false);
    }

    // C/D layout: VGPR j -> M = 8*hi + j, N = lane%16
    const int mbase = tileM * 16 + hi * 8;
    const int ncol  = tileN * 16 + l16;
    #pragma unroll
    for (int j = 0; j < 8; j++)
        O[(mbase + j) * DIM + ncol] = c[j];
}

// ---------------------------------------------------------------------------
// 4) Edge pass 1: e = sum_c leakyrelu(xl[src]+xr[dst]) * att ; segment max
// ---------------------------------------------------------------------------
__global__ void edge_score(const int* __restrict__ src,
                           const int* __restrict__ dst,
                           const float* __restrict__ xl,
                           const float* __restrict__ xr,
                           const float* __restrict__ att,   // [H][C] this layer
                           float* __restrict__ ea,          // [E][H]
                           float* __restrict__ m) {         // [N][H]
    int idx = blockIdx.x * blockDim.x + threadIdx.x;        // E * H
    if (idx >= N_EDGES * HEADS) return;
    int e  = idx >> 3;
    int hh = idx & 7;
    int s = src[e], d = dst[e];
    const float4* pl = (const float4*)(xl + s * DIM + hh * CHAN);
    const float4* pr = (const float4*)(xr + d * DIM + hh * CHAN);
    const float4* pa = (const float4*)(att + hh * CHAN);
    float acc = 0.0f;
    #pragma unroll
    for (int i = 0; i < CHAN / 4; i++) {
        float4 a = pl[i], b = pr[i], w = pa[i];
        float v;
        v = a.x + b.x; v = v > 0.0f ? v : SLOPE * v; acc += v * w.x;
        v = a.y + b.y; v = v > 0.0f ? v : SLOPE * v; acc += v * w.y;
        v = a.z + b.z; v = v > 0.0f ? v : SLOPE * v; acc += v * w.z;
        v = a.w + b.w; v = v > 0.0f ? v : SLOPE * v; acc += v * w.w;
    }
    ea[idx] = acc;
    atomic_max_f32(&m[d * HEADS + hh], acc);
}

// ---------------------------------------------------------------------------
// 5) Edge pass 2: a = exp(e - m[dst]); denom = segment_sum(a)
// ---------------------------------------------------------------------------
__global__ void edge_exp(const int* __restrict__ dst,
                         const float* __restrict__ m,
                         float* __restrict__ ea,
                         float* __restrict__ denom) {
    int idx = blockIdx.x * blockDim.x + threadIdx.x;        // E * H
    if (idx >= N_EDGES * HEADS) return;
    int e  = idx >> 3;
    int hh = idx & 7;
    int d = dst[e];
    float a = __expf(ea[idx] - m[d * HEADS + hh]);
    ea[idx] = a;
    atomicAdd(&denom[d * HEADS + hh], a);
}

// ---------------------------------------------------------------------------
// 6) Edge pass 3: alpha = a/(denom+eps); acc[dst] += alpha * xl[src]
// ---------------------------------------------------------------------------
__global__ void edge_scatter(const int* __restrict__ src,
                             const int* __restrict__ dst,
                             const float* __restrict__ ea,
                             const float* __restrict__ denom,
                             const float* __restrict__ xl,
                             float* __restrict__ acc) {
    int idx = blockIdx.x * blockDim.x + threadIdx.x;        // E * H
    if (idx >= N_EDGES * HEADS) return;
    int e  = idx >> 3;
    int hh = idx & 7;
    int s = src[e], d = dst[e];
    float alpha = ea[idx] / (denom[d * HEADS + hh] + 1e-16f);
    const float4* pl = (const float4*)(xl + s * DIM + hh * CHAN);
    float* pd = acc + d * DIM + hh * CHAN;
    #pragma unroll
    for (int i = 0; i < CHAN / 4; i++) {
        float4 a = pl[i];
        atomicAdd(pd + 4 * i + 0, alpha * a.x);
        atomicAdd(pd + 4 * i + 1, alpha * a.y);
        atomicAdd(pd + 4 * i + 2, alpha * a.z);
        atomicAdd(pd + 4 * i + 3, alpha * a.w);
    }
}

// ---------------------------------------------------------------------------
// 7) Node update: h += acc + bconv; refresh f16 copy for next layer's GEMM
// ---------------------------------------------------------------------------
__global__ void node_update(float* __restrict__ h,
                            const float* __restrict__ acc,
                            const float* __restrict__ bconv,  // [D] this layer
                            _Float16* __restrict__ hf) {
    int idx = blockIdx.x * blockDim.x + threadIdx.x;        // N * D
    if (idx >= N_NODES * DIM) return;
    float v = h[idx] + acc[idx] + bconv[idx & 255];
    h[idx]  = v;
    hf[idx] = (_Float16)v;
}

// ---------------------------------------------------------------------------
// 8) Pooling
// ---------------------------------------------------------------------------
__global__ void pool_init(float* __restrict__ sums, float* __restrict__ cnt) {
    int idx = blockIdx.x * blockDim.x + threadIdx.x;
    if (idx < GRAPHS * DIM) sums[idx] = 0.0f;
    if (idx < GRAPHS) cnt[idx] = 0.0f;
}

__global__ void pool(const float* __restrict__ h,
                     const int* __restrict__ batch,
                     float* __restrict__ sums,
                     float* __restrict__ cnt) {
    int idx = blockIdx.x * blockDim.x + threadIdx.x;        // N * D
    if (idx >= N_NODES * DIM) return;
    int n = idx >> 8, d = idx & 255;
    int g = batch[n];
    atomicAdd(&sums[g * DIM + d], h[idx]);
    if (d == 0) atomicAdd(&cnt[g], 1.0f);
}

// ---------------------------------------------------------------------------
// 9) Prediction head: out[g] = (sums[g]/max(cnt,1)) . Wpred + bpred
// ---------------------------------------------------------------------------
__global__ void predict(const float* __restrict__ sums,
                        const float* __restrict__ cnt,
                        const float* __restrict__ Wpred,
                        const float* __restrict__ bpred,
                        float* __restrict__ out) {
    __shared__ float red[DIM];
    int g = blockIdx.x;                                     // 0..15
    int t = threadIdx.x;                                    // 0..255
    float c = cnt[g]; c = c > 1.0f ? c : 1.0f;
    red[t] = (sums[g * DIM + t] / c) * Wpred[t];
    __syncthreads();
    for (int s = DIM / 2; s > 0; s >>= 1) {
        if (t < s) red[t] += red[t + s];
        __syncthreads();
    }
    if (t == 0) out[g] = red[0] + bpred[0];
}

// ---------------------------------------------------------------------------
// launch
// ---------------------------------------------------------------------------
extern "C" void kernel_launch(void* const* d_in, const int* in_sizes, int n_in,
                              void* d_out, int out_size, void* d_ws, size_t ws_size,
                              hipStream_t stream) {
    const float* x     = (const float*)d_in[0];
    const int*   ei    = (const int*)  d_in[1];   // [2][E]
    const int*   batch = (const int*)  d_in[2];
    const float* Wp    = (const float*)d_in[3];
    const float* bp    = (const float*)d_in[4];
    const float* Wl    = (const float*)d_in[5];   // [L][256][256]
    const float* Wr    = (const float*)d_in[6];
    const float* att   = (const float*)d_in[7];   // [L][H][C]
    const float* bconv = (const float*)d_in[8];   // [L][256]
    const float* Wpred = (const float*)d_in[9];
    const float* bpred = (const float*)d_in[10];
    float* out = (float*)d_out;

    const int* src = ei;
    const int* dst = ei + N_EDGES;

    // workspace layout
    const size_t ND = (size_t)N_NODES * DIM;   // 2,560,000
    const size_t EH = (size_t)N_EDGES * HEADS; // 2,560,000
    const size_t NH = (size_t)N_NODES * HEADS; //    80,000
    float* h      = (float*)d_ws;
    float* acc    = h + ND;
    float* xl     = acc + ND;
    float* xr     = xl + ND;
    float* ea     = xr + ND;
    float* m      = ea + EH;
    float* denom  = m + NH;
    float* sums   = denom + NH;
    float* cnt    = sums + GRAPHS * DIM;
    _Float16* hf  = (_Float16*)(cnt + 64);          // 16B aligned
    _Float16* Wt  = hf + ND;                        // 4 x 256 x 256 halves

    const int B = 256;
    const int gridND = (int)((ND + B - 1) / B);     // 10000
    const int gridEH = (int)((EH + B - 1) / B);     // 10000

    prep_weights<<<(4 * DIM * DIM + B - 1) / B, B, 0, stream>>>(Wl, Wr, Wt);
    project<<<gridND, B, 0, stream>>>(x, Wp, bp, h, hf);

    for (int l = 0; l < 2; l++) {
        layer_init<<<gridND, B, 0, stream>>>(acc, denom, m);
        gemm_wmma<<<dim3(N_NODES / 16, 2, 2), B, 0, stream>>>(
            hf, Wt + (2 * l + 0) * DIM * DIM, Wt + (2 * l + 1) * DIM * DIM,
            xl, xr);
        edge_score<<<gridEH, B, 0, stream>>>(src, dst, xl, xr,
                                             att + l * HEADS * CHAN, ea, m);
        edge_exp<<<gridEH, B, 0, stream>>>(dst, m, ea, denom);
        edge_scatter<<<gridEH, B, 0, stream>>>(src, dst, ea, denom, xl, acc);
        node_update<<<gridND, B, 0, stream>>>(h, acc, bconv + l * DIM, hf);
    }

    pool_init<<<(GRAPHS * DIM + B - 1) / B, B, 0, stream>>>(sums, cnt);
    pool<<<gridND, B, 0, stream>>>(h, batch, sums, cnt);
    predict<<<GRAPHS, DIM, 0, stream>>>(sums, cnt, Wpred, bpred, out);
}
